// TensorProductConvLayer_88776974008583
// MI455X (gfx1250) — compile-verified, matrix operand order checked by reading, not run on hardware
//
#include <hip/hip_runtime.h>
#include <hip/hip_bf16.h>

typedef __bf16 bf16_t;
typedef __attribute__((ext_vector_type(16))) __bf16 v16bf;
typedef __attribute__((ext_vector_type(8)))  float  v8f;

#define NS 48
#define NV 10
#define IN_SIZE 156            // 48 + 30 + 30 + 48
#define HIDDEN 96
#define NEDGEF 96
#define WNUMEL 6928            // 58*48 + 68*10 + 68*10 + 58*48
#define N_NODES 10000
#define N_EDGES 50000
#define TILE_E 32              // edges per workgroup (two 16-row WMMA tiles)
#define NTILES 433             // 6928 / 16
#define NBLOCKS ((N_EDGES + TILE_E - 1) / TILE_E)

#define INV_S3 0.57735026919f
#define INV_S2 0.70710678119f
#define INV_SQ58 0.13130643286f   // 1/sqrt(58)
#define INV_SQ68 0.12126781252f   // 1/sqrt(68)

// Build a 16x32 (or 32x16) bf16 WMMA operand fragment per CDNA5 ISA layout:
// lane in [0,15]: elems 0..7 -> K=kb+0..7,  elems 8..15 -> K=kb+16..23
// lane in [16,31]: elems 0..7 -> K=kb+8..15, elems 8..15 -> K=kb+24..31
__device__ inline v16bf make_frag(const bf16_t* row, int kb, int kh) {
  v16bf r;
  const bf16_t* p0 = row + kb + kh * 8;
  const bf16_t* p1 = p0 + 16;
#pragma unroll
  for (int j = 0; j < 8; ++j) { r[j] = p0[j]; r[j + 8] = p1[j]; }
  return r;
}

__device__ inline v16bf make_frag_f32(const float* row, int kb, int kh) {
  v16bf r;
  const float* p0 = row + kb + kh * 8;
  const float* p1 = p0 + 16;
#pragma unroll
  for (int j = 0; j < 8; ++j) { r[j] = (bf16_t)p0[j]; r[j + 8] = (bf16_t)p1[j]; }
  return r;
}

// ---------------- kernel 0: transpose + convert weights to bf16 ----------------
__global__ void convert_weights(const float* __restrict__ W1,
                                const float* __restrict__ W2,
                                bf16_t* __restrict__ W1T,
                                bf16_t* __restrict__ W2T) {
  int idx = blockIdx.x * 256 + threadIdx.x;
  if (idx < 96 * 96) {                    // W1T[n][k] = W1[k][n]
    int n = idx / 96, k = idx - n * 96;
    W1T[idx] = (bf16_t)W1[k * 96 + n];
  }
  int idx2 = idx - 96 * 96;
  if (idx2 >= 0 && idx2 < WNUMEL * 96) {  // W2T[n][k] = W2[k][n]
    int n = idx2 / 96, k = idx2 - n * 96;
    W2T[idx2] = (bf16_t)W2[(size_t)k * WNUMEL + n];
  }
}

// ---------------- kernel 1: fused h-GEMM + w-GEMM + tensor product + scatter ----------------
__global__ __launch_bounds__(256)
void tpconv_fused(const float* __restrict__ node_attr,
                  const int*   __restrict__ edge_index,
                  const float* __restrict__ edge_attr,
                  const float* __restrict__ edge_sh,
                  const bf16_t* __restrict__ W1T,
                  const float* __restrict__ b1,
                  const bf16_t* __restrict__ W2T,
                  const float* __restrict__ b2,
                  float* __restrict__ seg,
                  float* __restrict__ cnt) {
  __shared__ bf16_t h_lds[TILE_E][96];          //  6 KB
  __shared__ float x0e[TILE_E][58];             //  7.3 KB
  __shared__ float x1o[TILE_E][68 * 3];         // 25.5 KB
  __shared__ float x1e[TILE_E][68 * 3];         // 25.5 KB
  __shared__ float x0o[TILE_E][58];             //  7.3 KB
  __shared__ float accb[TILE_E][IN_SIZE];       // 19.5 KB   (total ~93 KB)

  const int t     = threadIdx.x;
  const int lane  = t & 31;
  const int wave  = t >> 5;      // 0..7
  const int m     = lane & 15;   // row inside 16x16 tile
  const int kh    = lane >> 4;   // which K half this lane supplies
  const int eBase = blockIdx.x * TILE_E;
  const int nE    = min(TILE_E, N_EDGES - eBase);   // valid edges in this tile

  // ---- zero per-edge output accumulator ----
  for (int p = t; p < TILE_E * IN_SIZE; p += 256) ((float*)accb)[p] = 0.0f;

  // ---- build equivariant input features for 32 edges (8 workers/edge) ----
  {
    const int e = t >> 3;          // 0..31
    const int s = t & 7;           // worker
    if (e < nE) {
      const int eg = eBase + e;
      const int dst = edge_index[N_EDGES + eg];
      const float* na = node_attr + (size_t)dst * IN_SIZE;
      const float sh0 = edge_sh[eg * 4 + 0];
      const float s1 = edge_sh[eg * 4 + 1];
      const float s2 = edge_sh[eg * 4 + 2];
      const float s3 = edge_sh[eg * 4 + 3];

      for (int j = s; j < 58; j += 8) {
        if (j < 48) x0e[e][j] = na[j] * sh0;
        else {
          int v = j - 48;
          x0e[e][j] = (na[48 + v * 3] * s1 + na[48 + v * 3 + 1] * s2 +
                       na[48 + v * 3 + 2] * s3) * INV_S3;
        }
      }
      for (int i = s; i < 68; i += 8) {
        float r0, r1, r2;
        if (i < 48)      { float f = na[i]; r0 = f * s1; r1 = f * s2; r2 = f * s3; }
        else if (i < 58) { int v = i - 48; r0 = na[48 + v * 3] * sh0;
                           r1 = na[48 + v * 3 + 1] * sh0; r2 = na[48 + v * 3 + 2] * sh0; }
        else {           // cross(in_1e[v], sh1o) * inv_s2
          int v = i - 58;
          float ax = na[78 + v * 3], ay = na[78 + v * 3 + 1], az = na[78 + v * 3 + 2];
          r0 = (ay * s3 - az * s2) * INV_S2;
          r1 = (az * s1 - ax * s3) * INV_S2;
          r2 = (ax * s2 - ay * s1) * INV_S2;
        }
        x1o[e][i * 3 + 0] = r0; x1o[e][i * 3 + 1] = r1; x1o[e][i * 3 + 2] = r2;
      }
      for (int i = s; i < 68; i += 8) {
        float r0, r1, r2;
        if (i < 10) {    // cross(in_1o[v], sh1o) * inv_s2
          int v = i;
          float ax = na[48 + v * 3], ay = na[48 + v * 3 + 1], az = na[48 + v * 3 + 2];
          r0 = (ay * s3 - az * s2) * INV_S2;
          r1 = (az * s1 - ax * s3) * INV_S2;
          r2 = (ax * s2 - ay * s1) * INV_S2;
        } else if (i < 20) { int v = i - 10; r0 = na[78 + v * 3] * sh0;
                             r1 = na[78 + v * 3 + 1] * sh0; r2 = na[78 + v * 3 + 2] * sh0; }
        else             { float f = na[108 + (i - 20)]; r0 = f * s1; r1 = f * s2; r2 = f * s3; }
        x1e[e][i * 3 + 0] = r0; x1e[e][i * 3 + 1] = r1; x1e[e][i * 3 + 2] = r2;
      }
      for (int j = s; j < 58; j += 8) {
        if (j < 10) {
          int v = j;
          x0o[e][j] = (na[78 + v * 3] * s1 + na[78 + v * 3 + 1] * s2 +
                       na[78 + v * 3 + 2] * s3) * INV_S3;
        } else x0o[e][j] = na[108 + (j - 10)] * sh0;
      }
    }
  }

  // ---- h = relu(edge_attr @ W1 + b1) for 32 edges: 2 row-tiles x 6 n-tiles ----
  for (int j = wave; j < 12; j += 8) {
    const int rt = j / 6;               // row tile 0/1
    const int nt = j - rt * 6;          // n tile 0..5
    int eRow = eBase + rt * 16 + m;
    if (eRow >= N_EDGES) eRow = N_EDGES - 1;   // clamp (result unused)
    const float* arow = edge_attr + (size_t)eRow * NEDGEF;
    v16bf a0 = make_frag_f32(arow, 0, kh);
    v16bf a1 = make_frag_f32(arow, 32, kh);
    v16bf a2 = make_frag_f32(arow, 64, kh);
    const int n = nt * 16 + m;
    const bf16_t* brow = W1T + n * 96;
    v16bf bb0 = make_frag(brow, 0, kh);
    v16bf bb1 = make_frag(brow, 32, kh);
    v16bf bb2 = make_frag(brow, 64, kh);
    const float bias = b1[n];
    v8f c;
#pragma unroll
    for (int r = 0; r < 8; ++r) c[r] = bias;
    c = __builtin_amdgcn_wmma_f32_16x16x32_bf16(false, a0, false, bb0, (short)0, c, false, false);
    c = __builtin_amdgcn_wmma_f32_16x16x32_bf16(false, a1, false, bb1, (short)0, c, false, false);
    c = __builtin_amdgcn_wmma_f32_16x16x32_bf16(false, a2, false, bb2, (short)0, c, false, false);
#pragma unroll
    for (int r = 0; r < 8; ++r)
      h_lds[rt * 16 + r + 8 * kh][n] = (bf16_t)fmaxf(c[r], 0.0f);
  }
  __syncthreads();

  // ---- A fragments (two 16-edge h tiles) reused for all 433 N-tiles ----
  const bf16_t* hrow0 = &h_lds[m][0];
  const bf16_t* hrow1 = &h_lds[16 + m][0];
  v16bf hA0[3], hA1[3];
#pragma unroll
  for (int kk = 0; kk < 3; ++kk) {
    hA0[kk] = make_frag(hrow0, kk * 32, kh);
    hA1[kk] = make_frag(hrow1, kk * 32, kh);
  }

  // ---- sweep W2 columns: each B fragment feeds TWO WMMAs (32 edge rows) ----
  for (int nt = wave; nt < NTILES; nt += 8) {
    const int n = nt * 16 + m;                 // this lane's weight column
    const bf16_t* brow = W2T + (size_t)n * 96;
    if (n + 128 < WNUMEL) __builtin_prefetch(W2T + (size_t)(n + 128) * 96, 0, 1);
    v16bf wb0 = make_frag(brow, 0, kh);
    v16bf wb1 = make_frag(brow, 32, kh);
    v16bf wb2 = make_frag(brow, 64, kh);
    const float bias = b2[n];
    v8f c0, c1;
#pragma unroll
    for (int r = 0; r < 8; ++r) { c0[r] = bias; c1[r] = bias; }
    c0 = __builtin_amdgcn_wmma_f32_16x16x32_bf16(false, hA0[0], false, wb0, (short)0, c0, false, false);
    c0 = __builtin_amdgcn_wmma_f32_16x16x32_bf16(false, hA0[1], false, wb1, (short)0, c0, false, false);
    c0 = __builtin_amdgcn_wmma_f32_16x16x32_bf16(false, hA0[2], false, wb2, (short)0, c0, false, false);
    c1 = __builtin_amdgcn_wmma_f32_16x16x32_bf16(false, hA1[0], false, wb0, (short)0, c1, false, false);
    c1 = __builtin_amdgcn_wmma_f32_16x16x32_bf16(false, hA1[1], false, wb1, (short)0, c1, false, false);
    c1 = __builtin_amdgcn_wmma_f32_16x16x32_bf16(false, hA1[2], false, wb2, (short)0, c1, false, false);

    // tensor-product consumption: column n -> (input idx, output idx, scale)
    if (n < 2784) {                            // 0e block: 58 x 48
      const int i = n / 48, o = n - i * 48;
#pragma unroll
      for (int r = 0; r < 8; ++r) {
        const int e0 = r + 8 * kh, e1 = 16 + e0;
        atomicAdd(&accb[e0][o], x0e[e0][i] * c0[r] * INV_SQ58);
        atomicAdd(&accb[e1][o], x0e[e1][i] * c1[r] * INV_SQ58);
      }
    } else if (n < 3464) {                     // 1o block: 68 x 10, 3 components
      const int mm = n - 2784; const int i = mm / 10, o = mm - i * 10;
#pragma unroll
      for (int r = 0; r < 8; ++r) {
        const int e0 = r + 8 * kh, e1 = 16 + e0;
        const float w0 = c0[r] * INV_SQ68, w1 = c1[r] * INV_SQ68;
#pragma unroll
        for (int cc = 0; cc < 3; ++cc) {
          atomicAdd(&accb[e0][48 + o * 3 + cc], x1o[e0][i * 3 + cc] * w0);
          atomicAdd(&accb[e1][48 + o * 3 + cc], x1o[e1][i * 3 + cc] * w1);
        }
      }
    } else if (n < 4144) {                     // 1e block: 68 x 10, 3 components
      const int mm = n - 3464; const int i = mm / 10, o = mm - i * 10;
#pragma unroll
      for (int r = 0; r < 8; ++r) {
        const int e0 = r + 8 * kh, e1 = 16 + e0;
        const float w0 = c0[r] * INV_SQ68, w1 = c1[r] * INV_SQ68;
#pragma unroll
        for (int cc = 0; cc < 3; ++cc) {
          atomicAdd(&accb[e0][78 + o * 3 + cc], x1e[e0][i * 3 + cc] * w0);
          atomicAdd(&accb[e1][78 + o * 3 + cc], x1e[e1][i * 3 + cc] * w1);
        }
      }
    } else {                                   // 0o block: 58 x 48
      const int mm = n - 4144; const int i = mm / 48, o = mm - i * 48;
#pragma unroll
      for (int r = 0; r < 8; ++r) {
        const int e0 = r + 8 * kh, e1 = 16 + e0;
        atomicAdd(&accb[e0][108 + o], x0o[e0][i] * c0[r] * INV_SQ58);
        atomicAdd(&accb[e1][108 + o], x0o[e1][i] * c1[r] * INV_SQ58);
      }
    }
  }
  __syncthreads();

  // ---- scatter per-edge rows into segment sums (global atomics) ----
  for (int p = t; p < nE * IN_SIZE; p += 256) {
    const int e = p / IN_SIZE, j = p - e * IN_SIZE;
    const int src = edge_index[eBase + e];
    atomicAdd(&seg[(size_t)src * IN_SIZE + j], accb[e][j]);
  }
  if (t < nE) {
    const int src = edge_index[eBase + t];
    atomicAdd(&cnt[src], 1.0f);
  }
}

// ---------------- kernel 2: mean + residual ----------------
__global__ void finalize(const float* __restrict__ seg,
                         const float* __restrict__ cnt,
                         const float* __restrict__ node_attr,
                         float* __restrict__ out) {
  int idx = blockIdx.x * 256 + threadIdx.x;
  if (idx >= N_NODES * IN_SIZE) return;
  int node = idx / IN_SIZE;
  out[idx] = seg[idx] / fmaxf(cnt[node], 1.0f) + node_attr[idx];
}

extern "C" void kernel_launch(void* const* d_in, const int* in_sizes, int n_in,
                              void* d_out, int out_size, void* d_ws, size_t ws_size,
                              hipStream_t stream) {
  const float* node_attr  = (const float*)d_in[0];
  const int*   edge_index = (const int*)  d_in[1];
  const float* edge_attr  = (const float*)d_in[2];
  const float* edge_sh    = (const float*)d_in[3];
  const float* W1         = (const float*)d_in[4];
  const float* b1         = (const float*)d_in[5];
  const float* W2         = (const float*)d_in[6];
  const float* b2         = (const float*)d_in[7];
  float* out = (float*)d_out;

  char* ws = (char*)d_ws;
  bf16_t* W1T = (bf16_t*)ws;                              // 96*96*2      = 18432 B
  bf16_t* W2T = (bf16_t*)(ws + 18432);                    // 6928*96*2    = 1330176 B
  float*  seg = (float*)(ws + 18432 + 1330176);           // 10000*156*4
  float*  cnt = seg + (size_t)N_NODES * IN_SIZE;          // 10000*4

  hipMemsetAsync(seg, 0, (size_t)(N_NODES * IN_SIZE + N_NODES) * sizeof(float), stream);

  {
    int total = 96 * 96 + WNUMEL * 96;
    convert_weights<<<(total + 255) / 256, 256, 0, stream>>>(W1, W2, W1T, W2T);
  }
  tpconv_fused<<<NBLOCKS, 256, 0, stream>>>(
      node_attr, edge_index, edge_attr, edge_sh, W1T, b1, W2T, b2, seg, cnt);
  {
    int total = N_NODES * IN_SIZE;
    finalize<<<(total + 255) / 256, 256, 0, stream>>>(seg, cnt, node_attr, out);
  }
}
